// PraxisExpert_30915174596851
// MI455X (gfx1250) — matrix operand order repaired and checked by reading, not compiled, not run on hardware
//
#include <hip/hip_runtime.h>

typedef __attribute__((ext_vector_type(16))) __bf16 v16bf;
typedef __attribute__((ext_vector_type(8)))  float  v8f;
typedef __attribute__((ext_vector_type(4)))  int    v4i;

typedef __attribute__((address_space(1))) v4i* g4p;   // global 16B chunk ptr
typedef __attribute__((address_space(3))) v4i* l4p;   // LDS 16B chunk ptr

#ifndef __has_builtin
#define __has_builtin(x) 0
#endif

#if __has_builtin(__builtin_amdgcn_global_load_async_to_lds_b128) && \
    __has_builtin(__builtin_amdgcn_s_wait_asynccnt)
#define USE_ASYNC_LDS 1
#else
#define USE_ASYNC_LDS 0
#endif

#define B_DIM   2
#define S_DIM   2048
#define D_DIM   1024
#define H_DIM   4096
#define E_DIM   8
#define K_SLOTS 2
#define T_TOK   (B_DIM * S_DIM)          // 4096 tokens
#define NPAIR   (T_TOK * K_SLOTS)        // 8192 (token,slot) pairs
#define CAP     NPAIR                    // per-expert list capacity (worst case)
#define TM      16                       // token tile (M) per workgroup
#define XS      (D_DIM + 16)             // LDS row stride, X tile (bf16): 520 dw % 64 = 8 banks/row
#define HS      (H_DIM + 16)             // LDS row stride, hidden tile (bf16)

// ---------------------------------------------------------------- routing ---
__global__ void moe_route_kernel(const int* __restrict__ idx,
                                 int* __restrict__ cnt,
                                 int* __restrict__ lists) {
    int p = blockIdx.x * blockDim.x + threadIdx.x;
    if (p < NPAIR) {
        int e = idx[p];
        int pos = atomicAdd(&cnt[e], 1);
        lists[e * CAP + pos] = p;
    }
}

// ------------------------------------------------------------- helpers ------
// Branch-free tanh-approx gelu: tanh(u) = 1 - 2/(exp2(u*2*log2(e)) + 1).
// v_exp_f32 saturates to +inf / 0 at the extremes, so no range branches.
__device__ __forceinline__ float gelu_fast(float x) {
    float u = 0.7978845608028654f * (x + 0.044715f * x * x * x);
    float e = __builtin_amdgcn_exp2f(2.8853900817779268f * u);   // e^(2u)
    float t = 1.0f - 2.0f * __builtin_amdgcn_rcpf(e + 1.0f);
    return 0.5f * x * (1.0f + t);
}

// A fragment (16x32 bf16, MxK) from LDS. Per ISA 7.12.2: lanes 0-15 hold
// row=lane with K = kb+{0..7, 16..23}; lanes 16-31 hold row=lane-16 with
// K = kb+{8..15, 24..31}.
__device__ __forceinline__ v16bf load_a_lds(const __bf16* base, int stride, int kb, int lane) {
    const __bf16* rp = base + (lane & 15) * stride + kb + ((lane >> 4) << 3);
    v16bf a;
    __builtin_memcpy(&a, rp, 16);                                     // K offsets +0..7
    __builtin_memcpy(reinterpret_cast<char*>(&a) + 16, rp + 16, 16);  // K offsets +16..23
    return a;
}

// B fragment (32x16 bf16, KxN) from row-major fp32 weights (ld = row stride).
// Lanes 0-15: K=kb+{0..7,16..23} at col=lane; lanes 16-31: K=kb+{8..15,24..31} at col=lane-16.
// Element i corresponds to K = kblk + i (i<8) or kblk + i + 8 (i>=8). Also prefetches the
// next K-step's rows (global_prefetch_b8).
__device__ __forceinline__ v16bf load_b_f32(const float* __restrict__ W, int ld,
                                            int kb, int n0, int lane) {
    int col  = lane & 15;
    int kblk = kb + ((lane >> 4) << 3);
    const float* wp = W + (size_t)kblk * ld + n0 + col;
    __builtin_prefetch(wp + (size_t)32 * ld, 0, 3);
    v16bf b;
#pragma unroll
    for (int i = 0; i < 16; ++i) b[i] = (__bf16)wp[(size_t)(i + ((i >> 3) << 3)) * ld];
    return b;
}

// ------------------------------------------------------- fused expert FFN ---
// grid = (E, CAP/TM), block = 256 (8 waves). One 16-row (token,slot) tile per
// block: hidden = gelu(X@W1+b1) staged in LDS as bf16, then out = hidden@W2+b2.
__global__ __launch_bounds__(256) void moe_ffn_kernel(
        const float* __restrict__ X,   const float* __restrict__ W1,
        const float* __restrict__ b1,  const float* __restrict__ W2,
        const float* __restrict__ b2,  const int* __restrict__ cnt,
        const int* __restrict__ lists, float* __restrict__ out) {
    extern __shared__ char smem[];
    __bf16* Xl    = (__bf16*)smem;                                  // [TM][XS]
    __bf16* Hl    = (__bf16*)(smem + (size_t)TM * XS * 2);          // [TM][HS]
    int*    pairs = (int*)   (smem + (size_t)TM * XS * 2 + (size_t)TM * HS * 2);
    float*  Xf    = (float*)Hl;   // fp32 staging (64KB) aliases hidden buffer

    const int e  = blockIdx.x;
    const int nE = cnt[e];
    const int rowbase = blockIdx.y * TM;
    if (rowbase >= nE) return;

    const int tid  = threadIdx.x;
    const int lane = tid & 31;
    const int wv   = tid >> 5;

    if (tid < TM) {
        int r = rowbase + tid;
        pairs[tid] = (r < nE) ? lists[e * CAP + r] : -1;
    }
    __syncthreads();

#if USE_ASYNC_LDS
    // Async-copy gathered fp32 X rows into LDS staging (GLOBAL_LOAD_ASYNC_TO_LDS_B128,
    // ASYNCcnt-tracked), then convert to bf16. Invalid rows zeroed in conversion.
    for (int ch = tid; ch < TM * (D_DIM / 4); ch += 256) {
        int r  = ch >> 8;                 // / (D_DIM/4)
        int c4 = ch & (D_DIM / 4 - 1);
        int p  = pairs[r];
        if (p >= 0) {
            const float* g = X + (size_t)(p / K_SLOTS) * D_DIM + c4 * 4;
            float*       l = Xf + r * D_DIM + c4 * 4;
            __builtin_amdgcn_global_load_async_to_lds_b128(
                (g4p)(uintptr_t)g,                 // v4i addrspace(1)*
                (l4p)(unsigned)(uintptr_t)l,       // v4i addrspace(3)* (low 32 bits = LDS offset)
                0, 0);
        }
    }
    __builtin_amdgcn_s_wait_asynccnt(0);
    __syncthreads();
    for (int i = tid; i < TM * D_DIM; i += 256) {
        int r = i >> 10;
        int c = i & (D_DIM - 1);
        Xl[r * XS + c] = (__bf16)((pairs[r] >= 0) ? Xf[r * D_DIM + c] : 0.0f);
    }
#else
    for (int i = tid; i < TM * D_DIM; i += 256) {
        int r = i >> 10;
        int c = i & (D_DIM - 1);
        int p = pairs[r];
        float v = 0.0f;
        if (p >= 0) v = X[(size_t)(p / K_SLOTS) * D_DIM + c];
        Xl[r * XS + c] = (__bf16)v;
    }
#endif
    __syncthreads();

    const float* W1e = W1 + (size_t)e * D_DIM * H_DIM;
    const float* b1e = b1 + (size_t)e * H_DIM;
    const int col = lane & 15;
    const int hi  = lane >> 4;

    // Phase 1: hidden tile (16 x H) = gelu(X @ W1 + b1), bf16 into LDS.
    for (int nt = wv; nt < H_DIM / 16; nt += 8) {
        int n0 = nt * 16;
        float bv = b1e[n0 + col];
        v8f acc = {bv, bv, bv, bv, bv, bv, bv, bv};
        for (int kb = 0; kb < D_DIM; kb += 32) {
            v16bf a = load_a_lds(Xl, XS, kb, lane);
            v16bf b = load_b_f32(W1e, H_DIM, kb, n0, lane);
            acc = __builtin_amdgcn_wmma_f32_16x16x32_bf16(
                      false, a, false, b, (short)0, acc, false, false);
        }
#pragma unroll
        for (int j = 0; j < 8; ++j) {               // C layout: row = j + 8*hi
            Hl[(j + 8 * hi) * HS + n0 + col] = (__bf16)gelu_fast(acc[j]);
        }
    }
    __syncthreads();

    const float* W2e = W2 + (size_t)e * H_DIM * D_DIM;
    const float* b2e = b2 + (size_t)e * D_DIM;

    // Phase 2: out rows (16 x D) = hidden @ W2 + b2, scattered to out[pair].
    for (int nt = wv; nt < D_DIM / 16; nt += 8) {
        int n0 = nt * 16;
        float bv = b2e[n0 + col];
        v8f acc = {bv, bv, bv, bv, bv, bv, bv, bv};
        for (int kb = 0; kb < H_DIM; kb += 32) {
            v16bf a = load_a_lds(Hl, HS, kb, lane);
            v16bf b = load_b_f32(W2e, D_DIM, kb, n0, lane);
            acc = __builtin_amdgcn_wmma_f32_16x16x32_bf16(
                      false, a, false, b, (short)0, acc, false, false);
        }
#pragma unroll
        for (int j = 0; j < 8; ++j) {
            int p = pairs[j + 8 * hi];
            if (p >= 0) out[(size_t)p * D_DIM + n0 + col] = acc[j];
        }
    }
}

// ------------------------------------------------------------------ launch --
extern "C" void kernel_launch(void* const* d_in, const int* in_sizes, int n_in,
                              void* d_out, int out_size, void* d_ws, size_t ws_size,
                              hipStream_t stream) {
    const float* X   = (const float*)d_in[0];
    const int*   idx = (const int*)  d_in[1];
    const float* W1  = (const float*)d_in[2];
    const float* b1  = (const float*)d_in[3];
    const float* W2  = (const float*)d_in[4];
    const float* b2  = (const float*)d_in[5];
    float* out = (float*)d_out;

    int* cnt   = (int*)d_ws;                       // E counters
    int* lists = (int*)((char*)d_ws + 256);        // E * CAP pair ids

    (void)hipMemsetAsync(cnt, 0, E_DIM * sizeof(int), stream);
    moe_route_kernel<<<(NPAIR + 255) / 256, 256, 0, stream>>>(idx, cnt, lists);

    size_t lds_bytes = (size_t)TM * XS * 2 + (size_t)TM * HS * 2 + TM * sizeof(int);
    dim3 grid(E_DIM, CAP / TM);
    moe_ffn_kernel<<<grid, 256, lds_bytes, stream>>>(X, W1, b1, W2, b2, cnt, lists, out);
}
